// Block_46651934769382
// MI455X (gfx1250) — compile-verified, matrix operand order checked by reading, not run on hardware
//
#include <hip/hip_runtime.h>

typedef __attribute__((ext_vector_type(16))) _Float16 v16h;
typedef __attribute__((ext_vector_type(8)))  _Float16 v8h;
typedef __attribute__((ext_vector_type(8)))  float    v8f;

#define NHEAD 16
#define HDIM  64
#define SEQ   2048
#define BATCH 2
#define CDIM  1024
#define RDIM  512
#define FDIM  4096

static __device__ inline v8f vzero8() {
  v8f z;
#pragma unroll
  for (int i = 0; i < 8; ++i) z[i] = 0.f;
  return z;
}

// Load a 16x32 f16 WMMA fragment from a row-major matrix (ld = leading dim in
// halves). Matches the CDNA5 16-bit A/B fragment layout:
//   lane L: row = base + (L&15); elems 0-7 = k0 + ((L>>4)*8) + 0..7,
//           elems 8-15 = k0 + 16 + ((L>>4)*8) + 0..7.
static __device__ inline v16h load_frag16x32(const _Float16* __restrict__ base,
                                             int ld, int row, int k0, int lane) {
  const int lr = lane & 15;
  const int kb = (lane >> 4) << 3;
  const _Float16* p = base + (size_t)(row + lr) * ld + (k0 + kb);
  v8h lo = *(const v8h*)(p);
  v8h hi = *(const v8h*)(p + 16);
  v16h r;
#pragma unroll
  for (int i = 0; i < 8; ++i) { r[i] = lo[i]; r[i + 8] = hi[i]; }
  return r;
}

static __device__ inline v8f wmma_f16(v16h a, v16h b, v8f c) {
  return __builtin_amdgcn_wmma_f32_16x16x32_f16(false, a, false, b, (short)0, c,
                                                false, false);
}

// CDNA5 async copy global -> LDS (ASYNCcnt-tracked, no VGPR data round-trip).
// LDS dest offset: ISA 10.2 — LDS-aperture flat address maps to LDS via
// addr[31:0], so truncating a generic pointer to a __shared__ object yields
// the wave-relative LDS byte offset the instruction wants.
static __device__ inline void async_ld_b128(unsigned ldsoff, const _Float16* g) {
  asm volatile("global_load_async_to_lds_b128 %0, %1, off"
               :: "v"(ldsoff), "v"((unsigned long long)(uintptr_t)g)
               : "memory");
}
static __device__ inline void wait_async0() {
  asm volatile("s_wait_asynccnt 0" ::: "memory");
}
static __device__ inline unsigned lds_off(const void* p) {
  return (unsigned)(uintptr_t)p;
}

// ---------------------------------------------------------------------------
// Weight convert+transpose: W[K,N] (f32, row-major) -> Wt[N,K] (f16, row-major)
// ---------------------------------------------------------------------------
__global__ void transpose_to_f16(const float* __restrict__ W,
                                 _Float16* __restrict__ Wt, int K, int N) {
  int idx = blockIdx.x * blockDim.x + threadIdx.x;
  if (idx >= N * K) return;
  int n = idx / K, k = idx - n * K;
  Wt[idx] = (_Float16)W[(size_t)k * N + n];
}

// ---------------------------------------------------------------------------
// Row LayerNorm (C=1024), f32 in -> f16 out
// ---------------------------------------------------------------------------
__global__ __launch_bounds__(256) void layernorm_to_f16(
    const float* __restrict__ x, const float* __restrict__ g,
    const float* __restrict__ b, _Float16* __restrict__ out, int C) {
  __shared__ float red[256];
  const int row = blockIdx.x;
  const int tid = threadIdx.x;
  const float* xr = x + (size_t)row * C;
  float s = 0.f, ss = 0.f;
  for (int i = tid; i < C; i += 256) { float v = xr[i]; s += v; ss += v * v; }
  red[tid] = s; __syncthreads();
  for (int o = 128; o > 0; o >>= 1) {
    if (tid < o) red[tid] += red[tid + o];
    __syncthreads();
  }
  const float mu = red[0] / C;
  __syncthreads();
  red[tid] = ss; __syncthreads();
  for (int o = 128; o > 0; o >>= 1) {
    if (tid < o) red[tid] += red[tid + o];
    __syncthreads();
  }
  const float var = red[0] / C - mu * mu;
  const float rs = rsqrtf(var + 1e-5f);
  for (int i = tid; i < C; i += 256)
    out[(size_t)row * C + i] = (_Float16)((xr[i] - mu) * rs * g[i] + b[i]);
}

// ---------------------------------------------------------------------------
// WMMA GEMM: C[M,N] = A[M,K] * Bt[N,K]^T   (A,Bt f16 row-major, f32 accum)
// Block = 256 threads = 8 waves; block tile 128(M) x 64(N) x 32(K).
// Tiles double-buffered in LDS via global_load_async_to_lds_b128 (ASYNCcnt);
// fragments then come from ds_load_b128.  Wave tile 32x32 (4 WMMA / k-step).
// EPI: 0=f16 store  1=K/V^T scatter  2=Q scatter  3=f32+resid
//      4=f16 bias+relu  5=f32+bias+resid
// ---------------------------------------------------------------------------
template <int EPI>
__global__ __launch_bounds__(256) void gemm_wmma_kernel(
    const _Float16* __restrict__ A, const _Float16* __restrict__ Bt,
    int M, int N, int K,
    float* __restrict__ Cf, _Float16* __restrict__ Ch,
    const float* __restrict__ bias, const float* __restrict__ resid,
    _Float16* __restrict__ kOut, _Float16* __restrict__ vtOut) {
  __shared__ __align__(16) _Float16 As[2 * 128 * 32];  // 16 KB
  __shared__ __align__(16) _Float16 Bs[2 * 64 * 32];   //  8 KB
  const int tid = threadIdx.x;
  const int lane = tid & 31;
  const int wave = tid >> 5;
  const int lr = lane & 15;
  const int half = lane >> 4;
  const int m0b = blockIdx.y * 128;
  const int n0b = blockIdx.x * 64;
  const int wm = (wave & 3) * 32;   // wave M offset inside block tile
  const int wn = (wave >> 2) * 32;  // wave N offset inside block tile
  (void)M;

  // Per-thread async-copy lanes: 3 x b128 (16B) chunks per k-step.
  const int ar0 = tid >> 2;          // A rows 0..63
  const int ar1 = ar0 + 64;          // A rows 64..127
  const int ak = (tid & 3) * 8;      // 8-half column chunk
  const int br = tid >> 2;           // B rows 0..63
  const _Float16* ag0 = A + (size_t)(m0b + ar0) * K + ak;
  const _Float16* ag1 = A + (size_t)(m0b + ar1) * K + ak;
  const _Float16* bg = Bt + (size_t)(n0b + br) * K + ak;
  const unsigned al0 = lds_off(&As[ar0 * 32 + ak]);
  const unsigned al1 = lds_off(&As[ar1 * 32 + ak]);
  const unsigned bl = lds_off(&Bs[br * 32 + ak]);

  // Prologue: stage k=0 into buffer 0.
  async_ld_b128(al0, ag0);
  async_ld_b128(al1, ag1);
  async_ld_b128(bl, bg);

  v8f c00 = vzero8(), c01 = vzero8(), c10 = vzero8(), c11 = vzero8();
  int cur = 0;
  wait_async0();
  __syncthreads();

  for (int k = 0; k < K; k += 32) {
    const int nxt = cur ^ 1;
    if (k + 32 < K) {  // issue next tile's async copies (other buffer)
      async_ld_b128(al0 + nxt * 8192, ag0 + k + 32);
      async_ld_b128(al1 + nxt * 8192, ag1 + k + 32);
      async_ld_b128(bl + nxt * 4096, bg + k + 32);
    }
    const _Float16* as = &As[cur * 4096];
    const _Float16* bs = &Bs[cur * 2048];
    v16h a0 = load_frag16x32(as, 32, wm, 0, lane);
    v16h a1 = load_frag16x32(as, 32, wm + 16, 0, lane);
    v16h b0 = load_frag16x32(bs, 32, wn, 0, lane);
    v16h b1 = load_frag16x32(bs, 32, wn + 16, 0, lane);
    c00 = wmma_f16(a0, b0, c00);
    c01 = wmma_f16(a0, b1, c01);
    c10 = wmma_f16(a1, b0, c10);
    c11 = wmma_f16(a1, b1, c11);
    wait_async0();   // own async writes to the other buffer have landed
    __syncthreads(); // everyone's writes landed; everyone's reads of cur done
    cur = nxt;
  }

  const int m0 = m0b + wm, n0 = n0b + wn;
  v8f accs[4] = {c00, c01, c10, c11};
#pragma unroll
  for (int t = 0; t < 4; ++t) {
    const int rowB = m0 + (t >> 1) * 16 + half * 8;
    const int col = n0 + (t & 1) * 16 + lr;
#pragma unroll
    for (int r = 0; r < 8; ++r) {
      const int row = rowB + r;
      const float v = accs[t][r];
      if (EPI == 0) {
        Ch[(size_t)row * N + col] = (_Float16)v;
      } else if (EPI == 1) {  // kv: col<H*D -> K[b,h,t,d] ; else V^T[b,h,d,t]
        const int bb = row / SEQ, tt = row - bb * SEQ;
        if (col < NHEAD * HDIM) {
          const int h = col / HDIM, d = col - h * HDIM;
          kOut[(((size_t)bb * NHEAD + h) * SEQ + tt) * HDIM + d] = (_Float16)v;
        } else {
          const int cc = col - NHEAD * HDIM;
          const int h = cc / HDIM, d = cc - h * HDIM;
          vtOut[(((size_t)bb * NHEAD + h) * HDIM + d) * SEQ + tt] = (_Float16)v;
        }
      } else if (EPI == 2) {  // q -> Q[b,h,t,d]
        const int bb = row / SEQ, tt = row - bb * SEQ;
        const int h = col / HDIM, d = col - h * HDIM;
        Ch[(((size_t)bb * NHEAD + h) * SEQ + tt) * HDIM + d] = (_Float16)v;
      } else if (EPI == 3) {
        Cf[(size_t)row * N + col] = v + resid[(size_t)row * N + col];
      } else if (EPI == 4) {
        const float t2 = v + bias[col];
        Ch[(size_t)row * N + col] = (_Float16)(t2 > 0.f ? t2 : 0.f);
      } else {  // 5
        Cf[(size_t)row * N + col] = v + bias[col] + resid[(size_t)row * N + col];
      }
    }
  }
}

// ---------------------------------------------------------------------------
// Causal flash attention. 1 wave = 16 query rows; block = 4 waves (64 rows).
// Q,K in [b,h,t,d] f16; V in [b,h,d,t] f16 (transposed); out y[b,t,h*d] f16.
// 8 v_wmma per 32-key step (4 for QK^T, 4 for P*V).
// ---------------------------------------------------------------------------
__global__ __launch_bounds__(128) void attention_kernel(
    const _Float16* __restrict__ qf, const _Float16* __restrict__ kf,
    const _Float16* __restrict__ vtf, _Float16* __restrict__ y) {
  __shared__ __align__(16) _Float16 pbuf[4][16 * 32];
  const int lane = threadIdx.x & 31;
  const int wave = threadIdx.x >> 5;
  const int lr = lane & 15;
  const int half = lane >> 4;
  const int qs = blockIdx.x * 64 + wave * 16;
  const int bh = blockIdx.y;

  const _Float16* qp = qf + (size_t)bh * SEQ * HDIM;
  const _Float16* kp = kf + (size_t)bh * SEQ * HDIM;
  const _Float16* vp = vtf + (size_t)bh * HDIM * SEQ;

  const v16h aq0 = load_frag16x32(qp, HDIM, qs, 0, lane);
  const v16h aq1 = load_frag16x32(qp, HDIM, qs, 32, lane);

  float m[8], l[8];
  v8f acc[4];
#pragma unroll
  for (int r = 0; r < 8; ++r) { m[r] = -3.0e38f; l[r] = 0.f; }
#pragma unroll
  for (int d = 0; d < 4; ++d) acc[d] = vzero8();

  const float scale = 0.125f;  // D^-0.5, D=64
  const int kend = qs + 16;    // causal: keys <= qs+15
  _Float16* pb = pbuf[wave];

  for (int kt = 0; kt < kend; kt += 32) {
    // S = Q K^T for keys [kt, kt+32)
    v8f s0 = vzero8(), s1 = vzero8();
    s0 = wmma_f16(aq0, load_frag16x32(kp, HDIM, kt, 0, lane), s0);
    s0 = wmma_f16(aq1, load_frag16x32(kp, HDIM, kt, 32, lane), s0);
    s1 = wmma_f16(aq0, load_frag16x32(kp, HDIM, kt + 16, 0, lane), s1);
    s1 = wmma_f16(aq1, load_frag16x32(kp, HDIM, kt + 16, 32, lane), s1);

    const int key0 = kt + lr, key1 = kt + 16 + lr;
#pragma unroll
    for (int r = 0; r < 8; ++r) {
      const int qrow = qs + half * 8 + r;
      float sv0 = (key0 <= qrow) ? s0[r] * scale : -1.0e30f;
      float sv1 = (key1 <= qrow) ? s1[r] * scale : -1.0e30f;
      float tm = fmaxf(sv0, sv1);
#pragma unroll
      for (int off = 1; off < 16; off <<= 1)
        tm = fmaxf(tm, __shfl_xor(tm, off, 32));
      const float mn = fmaxf(m[r], tm);
      const float sc = __expf(m[r] - mn);
      const float p0 = __expf(sv0 - mn);
      const float p1 = __expf(sv1 - mn);
      float rs = p0 + p1;
#pragma unroll
      for (int off = 1; off < 16; off <<= 1) rs += __shfl_xor(rs, off, 32);
      l[r] = l[r] * sc + rs;
      m[r] = mn;
#pragma unroll
      for (int d = 0; d < 4; ++d) acc[d][r] *= sc;
      // Stage P (C-layout -> row-major 16x32) into per-wave LDS tile.
      pb[(half * 8 + r) * 32 + lr] = (_Float16)p0;
      pb[(half * 8 + r) * 32 + 16 + lr] = (_Float16)p1;
    }
    // Same-wave LDS ops are in-order (DScnt); fence keeps compiler honest.
    asm volatile("s_wait_dscnt 0" ::: "memory");
    const v16h ap = load_frag16x32(pb, 32, 0, 0, lane);
#pragma unroll
    for (int d = 0; d < 4; ++d)
      acc[d] = wmma_f16(ap, load_frag16x32(vp, SEQ, d * 16, kt, lane), acc[d]);
  }

  const int bb = bh / NHEAD, h = bh - bb * NHEAD;
#pragma unroll
  for (int d = 0; d < 4; ++d) {
#pragma unroll
    for (int r = 0; r < 8; ++r) {
      const int q = qs + half * 8 + r;
      y[((size_t)bb * SEQ + q) * (NHEAD * HDIM) + h * HDIM + d * 16 + lr] =
          (_Float16)(acc[d][r] / l[r]);
    }
  }
}

// ---------------------------------------------------------------------------
extern "C" void kernel_launch(void* const* d_in, const int* in_sizes, int n_in,
                              void* d_out, int out_size, void* d_ws,
                              size_t ws_size, hipStream_t stream) {
  (void)in_sizes; (void)n_in; (void)out_size; (void)ws_size;
  const float* x     = (const float*)d_in[0];
  const float* ln1_g = (const float*)d_in[1];
  const float* ln1_b = (const float*)d_in[2];
  const float* W_kvd = (const float*)d_in[3];
  const float* W_kvu = (const float*)d_in[4];
  const float* W_q   = (const float*)d_in[5];
  const float* W_o   = (const float*)d_in[6];
  const float* ln2_g = (const float*)d_in[7];
  const float* ln2_b = (const float*)d_in[8];
  const float* W_ff1 = (const float*)d_in[9];
  const float* b_ff1 = (const float*)d_in[10];
  const float* W_ff2 = (const float*)d_in[11];
  const float* b_ff2 = (const float*)d_in[12];
  float* out = (float*)d_out;

  const int BT = BATCH * SEQ;  // 4096
  char* ws = (char*)d_ws;
  size_t off = 0;
  auto alloc = [&](size_t bytes) {
    void* p = ws + off;
    off = (off + bytes + 255) & ~(size_t)255;
    return p;
  };
  _Float16* h1    = (_Float16*)alloc((size_t)BT * CDIM * 2);
  _Float16* wtKvd = (_Float16*)alloc((size_t)RDIM * CDIM * 2);
  _Float16* wtKvu = (_Float16*)alloc((size_t)2 * CDIM * RDIM * 2);
  _Float16* wtQ   = (_Float16*)alloc((size_t)CDIM * CDIM * 2);
  _Float16* wtO   = (_Float16*)alloc((size_t)CDIM * CDIM * 2);
  _Float16* wtF1  = (_Float16*)alloc((size_t)FDIM * CDIM * 2);
  _Float16* wtF2  = (_Float16*)alloc((size_t)CDIM * FDIM * 2);
  _Float16* lat   = (_Float16*)alloc((size_t)BT * RDIM * 2);
  _Float16* kf    = (_Float16*)alloc((size_t)BT * CDIM * 2);
  _Float16* vtf   = (_Float16*)alloc((size_t)BT * CDIM * 2);
  _Float16* qf    = (_Float16*)alloc((size_t)BT * CDIM * 2);
  _Float16* yf    = (_Float16*)alloc((size_t)BT * CDIM * 2);
  float*    xatt  = (float*)alloc((size_t)BT * CDIM * 4);
  _Float16* h2    = (_Float16*)alloc((size_t)BT * CDIM * 2);
  _Float16* ff1   = (_Float16*)alloc((size_t)BT * FDIM * 2);

  auto tpose = [&](const float* W, _Float16* Wt, int K, int N) {
    const int tot = K * N;
    transpose_to_f16<<<(tot + 255) / 256, 256, 0, stream>>>(W, Wt, K, N);
  };
  tpose(W_kvd, wtKvd, CDIM, RDIM);
  tpose(W_kvu, wtKvu, RDIM, 2 * CDIM);
  tpose(W_q,   wtQ,   CDIM, CDIM);
  tpose(W_o,   wtO,   CDIM, CDIM);
  tpose(W_ff1, wtF1,  CDIM, FDIM);
  tpose(W_ff2, wtF2,  FDIM, CDIM);

  layernorm_to_f16<<<BT, 256, 0, stream>>>(x, ln1_g, ln1_b, h1, CDIM);

  const dim3 blk(256);
  auto gg = [](int M, int N) { return dim3(N / 64, M / 128); };

  // latent = h @ W_kv_down
  gemm_wmma_kernel<0><<<gg(BT, RDIM), blk, 0, stream>>>(
      h1, wtKvd, BT, RDIM, CDIM, nullptr, lat, nullptr, nullptr, nullptr, nullptr);
  // kv = latent @ W_kv_up  (scatter K[b,h,t,d], V^T[b,h,d,t])
  gemm_wmma_kernel<1><<<gg(BT, 2 * CDIM), blk, 0, stream>>>(
      lat, wtKvu, BT, 2 * CDIM, RDIM, nullptr, nullptr, nullptr, nullptr, kf, vtf);
  // q = h @ W_q  (scatter Q[b,h,t,d])
  gemm_wmma_kernel<2><<<gg(BT, CDIM), blk, 0, stream>>>(
      h1, wtQ, BT, CDIM, CDIM, nullptr, qf, nullptr, nullptr, nullptr, nullptr);

  attention_kernel<<<dim3(SEQ / 64, BATCH * NHEAD), 128, 0, stream>>>(qf, kf, vtf, yf);

  // x_attn = x + y @ W_o
  gemm_wmma_kernel<3><<<gg(BT, CDIM), blk, 0, stream>>>(
      yf, wtO, BT, CDIM, CDIM, xatt, nullptr, nullptr, x, nullptr, nullptr);

  layernorm_to_f16<<<BT, 256, 0, stream>>>(xatt, ln2_g, ln2_b, h2, CDIM);

  // ff = relu(h2 @ W_ff1 + b1)
  gemm_wmma_kernel<4><<<gg(BT, FDIM), blk, 0, stream>>>(
      h2, wtF1, BT, FDIM, CDIM, nullptr, ff1, b_ff1, nullptr, nullptr, nullptr);
  // out = x_attn + ff @ W_ff2 + b2
  gemm_wmma_kernel<5><<<gg(BT, CDIM), blk, 0, stream>>>(
      ff1, wtF2, BT, CDIM, FDIM, out, nullptr, b_ff2, xatt, nullptr, nullptr);
}
